// DeformableExtractor_79250736546600
// MI455X (gfx1250) — matrix operand order repaired and checked by reading, not compiled, not run on hardware
//
#include <hip/hip_runtime.h>
#include <hip/hip_bf16.h>
#include <math.h>

typedef __attribute__((ext_vector_type(16))) _Float16 v16h;
typedef __attribute__((ext_vector_type(8)))  float    v8f;
typedef unsigned int u32x4 __attribute__((ext_vector_type(4)));
typedef int          i32x8 __attribute__((ext_vector_type(8)));
typedef int          i32x4 __attribute__((ext_vector_type(4)));

#define Hh 128
#define Ww 128
#define HW 16384
#define Tt 5
#define Bb 2
#define Ff 64

// ---------------------------------------------------------------------------
// Pack fp32 weight matrix [Mvalid x Kdim] into f16 WMMA A-fragment layout:
// dst[((mt*Ksteps+ks)*32+lane)*16+i] = w[M= mt*16+(lane&15)][K= ks*32 + kmap]
// kmap (contiguous k order, matches B panel): K = (lane>>4)*16 + i
// ---------------------------------------------------------------------------
__global__ void pack_a_f16(const float* __restrict__ wsrc, _Float16* __restrict__ dst,
                           int Mtiles, int Ksteps, int Mvalid, int Kdim) {
    int idx = blockIdx.x * 256 + threadIdx.x;
    int total = Mtiles * Ksteps * 512;
    if (idx >= total) return;
    int i    = idx & 15;
    int lane = (idx >> 4) & 31;
    int ks   = (idx >> 9) % Ksteps;
    int mt   = idx / (512 * Ksteps);
    int hs   = lane >> 4;
    int M    = mt * 16 + (lane & 15);
    int K    = ks * 32 + hs * 16 + i;
    float v  = (M < Mvalid) ? wsrc[(size_t)M * Kdim + K] : 0.0f;
    dst[idx] = (_Float16)v;
}

// B-panel fragment-order index: K-index r (= ks*32+kk), pixel n (0..15).
// B layout (16-bit 32x16): lane = (kk&16)+n holds element i = kk&15.
__device__ __forceinline__ int bfrag_idx(int r, int n) {
    int ks = r >> 5;
    int kk = r & 31;
    return ks * 512 + ((kk & 16) + n) * 16 + (kk & 15);
}

// ---------------------------------------------------------------------------
// Stage 1: 3x3 SAME conv (Cin=3 -> F=64) + PReLU. Tiny K, pure VALU is fine.
// feats layout: [(b*T+j)*64 + f][HW]
// ---------------------------------------------------------------------------
__global__ void init_conv_prelu(const float* __restrict__ x, const float* __restrict__ w,
                                const float* __restrict__ bias, const float* __restrict__ pa,
                                float* __restrict__ feats) {
    int idx = blockIdx.x * 256 + threadIdx.x;
    if (idx >= Bb * Tt * Ff * HW) return;
    int pix = idx & (HW - 1);
    int f   = (idx >> 14) & 63;
    int bt  = idx >> 20;
    int y = pix >> 7, xw = pix & 127;
    float acc = bias[f];
    for (int c = 0; c < 3; ++c) {
#pragma unroll
        for (int tap = 0; tap < 9; ++tap) {
            int dy = tap / 3 - 1, dx = tap % 3 - 1;
            int yy = y + dy, xx = xw + dx;
            if (yy >= 0 && yy < Hh && xx >= 0 && xx < Ww)
                acc += x[((size_t)bt * 3 + c) * HW + yy * Ww + xx] * w[(f * 3 + c) * 9 + tap];
        }
    }
    float a = pa[0];
    feats[idx] = acc > 0.0f ? acc : a * acc;
}

// ---------------------------------------------------------------------------
// Stage 2: offset conv as implicit GEMM, M=27(->32), K=1152, N=16 pixels/block.
// 2 waves, each owns one 16-row M-tile; 36 K-steps of v_wmma_f32_16x16x32_f16.
// B panel stored in fragment order -> one 32B ds_load per lane per K-step.
// ---------------------------------------------------------------------------
__global__ void __launch_bounds__(64)
offset_conv_wmma(const float* __restrict__ feats, const _Float16* __restrict__ apack,
                 const float* __restrict__ offm_b, float* __restrict__ om) {
    __shared__ __align__(32) _Float16 bpanel[1152 * 16];   // 36 KB, fragment order
    int blk  = blockIdx.x;
    int jb   = blk >> 10;            // j*2 + b, 0..9
    int tile = blk & 1023;
    int j = jb >> 1, b = jb & 1;
    int pixbase = tile * 16;
    int y  = pixbase >> 7;           // whole 16-pixel tile sits on one row
    int x0 = pixbase & 127;
    const float* f0 = feats + (size_t)((b * Tt + 0) * Ff) * HW;
    const float* fj = feats + (size_t)((b * Tt + j) * Ff) * HW;
    int t = threadIdx.x;

    // Fill B panel (fragment order): value = extra[cin][y+dy][x+dx], r = cin*9+tap
    for (int e = t; e < 1152 * 16; e += 64) {
        int n   = e & 15;
        int r   = e >> 4;
        int cin = r / 9;
        int tap = r - cin * 9;
        int yy = y + tap / 3 - 1;
        int xx = x0 + n + tap % 3 - 1;
        float v = 0.0f;
        if (yy >= 0 && yy < Hh && xx >= 0 && xx < Ww) {
            const float* src = (cin < 64) ? (f0 + (size_t)cin * HW) : (fj + (size_t)(cin - 64) * HW);
            v = src[yy * Ww + xx];
        }
        bpanel[bfrag_idx(r, n)] = (_Float16)v;
    }
    __syncthreads();

    int wave = t >> 5;               // M-tile id
    int lane = t & 31;
    int hs = lane >> 4, n = lane & 15;
    int lofs = lane * 16;
    v8f acc = {};
    const _Float16* ap = apack + (size_t)wave * 36 * 512 + lofs;
#pragma unroll 4
    for (int ks = 0; ks < 36; ++ks) {
        v16h afrag = *(const v16h*)(ap + (size_t)ks * 512);
        v16h bfrag = *(const v16h*)(bpanel + ks * 512 + lofs);
        acc = __builtin_amdgcn_wmma_f32_16x16x32_f16(false, afrag, false, bfrag,
                                                     (short)0, acc, false, false);
    }

    int pix = pixbase + n;
#pragma unroll
    for (int vi = 0; vi < 8; ++vi) {
        int ch = wave * 16 + vi + hs * 8;      // C-layout: M = vgpr + (lane>=16)*8
        if (ch < 27) {
            float val = acc[vi] + offm_b[ch];
            if (ch >= 18) val = 1.0f / (1.0f + __expf(-val));   // fused sigmoid (mask)
            om[((size_t)jb * 27 + ch) * HW + pix] = val;
        }
    }
}

__device__ __forceinline__ float tap_read(const float* src, int yi, int xi) {
    if (yi < 0 || yi >= Hh || xi < 0 || xi >= Ww) return 0.0f;
    return src[yi * Ww + xi];
}

// ---------------------------------------------------------------------------
// Stage 3: deformable sampling + einsum GEMM. M=64 (4 waves x 1 M-tile each),
// K=576 (18 steps), N=16 pixels. The 27x16 offset/mask tile is DMA'd into LDS
// once per block by the Tensor Data Mover (it is otherwise re-read 64x from
// global), then the fill loop builds the fragment-order f16 B panel.
// ---------------------------------------------------------------------------
__global__ void __launch_bounds__(128)
dcn_wmma(const float* __restrict__ feats, const float* __restrict__ om,
         const _Float16* __restrict__ apack, const float* __restrict__ dcn_b,
         float* __restrict__ out) {
    __shared__ __align__(32) _Float16 bpanel[576 * 16];    // 18 KB, fragment order
    __shared__ __align__(16) float    omlds[27 * 16];      // TDM-staged o_y/o_x/mask tile
    int blk  = blockIdx.x;
    int jb   = blk >> 10;
    int tile = blk & 1023;
    int j = jb >> 1, b = jb & 1;
    int pixbase = tile * 16;
    int ybase = pixbase >> 7;
    int x0 = pixbase & 127;
    const float* fj  = feats + (size_t)((b * Tt + j) * Ff) * HW;
    const float* omp = om + (size_t)jb * 27 * HW;
    int t = threadIdx.x;

    // --- TDM: 2D tile load (27 rows x 16 px, row stride HW) global -> LDS ---
    if (t < 32) {                               // wave 0 issues; EXEC ignored by TDM
        unsigned long long ga = (unsigned long long)(size_t)(omp + pixbase);
        unsigned lds_addr = (unsigned)(size_t)(void*)omlds;   // generic addr low 32b = LDS offset
        u32x4 g0;
        g0[0] = 1u;                                            // count=1 (valid), user mode
        g0[1] = lds_addr;                                      // D# lds_addr
        g0[2] = (unsigned)(ga & 0xffffffffu);                  // global_addr[31:0]
        g0[3] = (unsigned)((ga >> 32) & 0x01ffffffu) | (2u << 30);  // addr[56:32], type=2
        i32x8 g1;
        g1[0] = (2 << 16);                                     // data_size = 4 bytes
        g1[1] = (int)((HW & 0xffff) << 16);                    // tensor_dim0[15:0]
        g1[2] = (int)((HW >> 16) | (27 << 16));                // tensor_dim0[31:16] | tensor_dim1 lo
        g1[3] = (int)(16 << 16);                               // tensor_dim1 hi | tile_dim0 = 16
        g1[4] = 27;                                            // tile_dim1 = 27, tile_dim2 = 0
        g1[5] = HW;                                            // tensor_dim0_stride[31:0]
        g1[6] = 0;                                             // stride0 hi | stride1 lo
        g1[7] = 0;
        i32x4 gz4 = {0, 0, 0, 0};                              // 2D tensor: groups 2/3 unused
        i32x8 gz8 = {0, 0, 0, 0, 0, 0, 0, 0};
        __builtin_amdgcn_tensor_load_to_lds(g0, g1, gz4, gz4, gz8, 0);
        __builtin_amdgcn_s_wait_tensorcnt(0);
    }
    __syncthreads();

    for (int e = t; e < 9216; e += 128) {    // 16 px * 9 taps * 64 ch
        int c    = e & 63;
        int pair = e >> 6;
        int n = pair / 9;
        int k = pair - n * 9;
        float oy = omlds[k * 16 + n];
        float ox = omlds[(9 + k) * 16 + n];
        float m  = omlds[(18 + k) * 16 + n];             // already sigmoided
        float py = (float)ybase + (float)(k / 3 - 1) + oy;
        float px = (float)(x0 + n) + (float)(k % 3 - 1) + ox;
        float y0f = floorf(py), x0f = floorf(px);
        float ly = py - y0f, lx = px - x0f;
        int iy = (int)y0f, ix = (int)x0f;
        const float* src = fj + (size_t)c * HW;
        float v00 = tap_read(src, iy,     ix);
        float v01 = tap_read(src, iy,     ix + 1);
        float v10 = tap_read(src, iy + 1, ix);
        float v11 = tap_read(src, iy + 1, ix + 1);
        float val = (1.0f - ly) * ((1.0f - lx) * v00 + lx * v01) +
                    ly          * ((1.0f - lx) * v10 + lx * v11);
        bpanel[bfrag_idx(c * 9 + k, n)] = (_Float16)(val * m);
    }
    __syncthreads();

    int wave = t >> 5;               // M-tile 0..3 -> output channels wave*16..+15
    int lane = t & 31;
    int hs = lane >> 4, n = lane & 15;
    int lofs = lane * 16;
    v8f acc = {};
    const _Float16* ap = apack + (size_t)wave * 18 * 512 + lofs;
#pragma unroll 6
    for (int ks = 0; ks < 18; ++ks) {
        v16h afrag = *(const v16h*)(ap + (size_t)ks * 512);
        v16h bfrag = *(const v16h*)(bpanel + ks * 512 + lofs);
        acc = __builtin_amdgcn_wmma_f32_16x16x32_f16(false, afrag, false, bfrag,
                                                     (short)0, acc, false, false);
    }

    int pix = pixbase + n;
#pragma unroll
    for (int vi = 0; vi < 8; ++vi) {
        int o = wave * 16 + vi + hs * 8;
        out[((size_t)jb * Ff + o) * HW + pix] = acc[vi] + dcn_b[o];
    }
}

// ---------------------------------------------------------------------------
// inputs: x, init_w, init_b, prelu_a, offm_w, offm_b, dcn_w, dcn_b
// out: (T,B,64,H,W) fp32.  ws: feats(40MB) | om(17MB) | apack_offm | apack_dcn
// ---------------------------------------------------------------------------
extern "C" void kernel_launch(void* const* d_in, const int* in_sizes, int n_in,
                              void* d_out, int out_size, void* d_ws, size_t ws_size,
                              hipStream_t stream) {
    const float* x       = (const float*)d_in[0];
    const float* init_w  = (const float*)d_in[1];
    const float* init_b  = (const float*)d_in[2];
    const float* prelu_a = (const float*)d_in[3];
    const float* offm_w  = (const float*)d_in[4];
    const float* offm_b  = (const float*)d_in[5];
    const float* dcn_w   = (const float*)d_in[6];
    const float* dcn_b   = (const float*)d_in[7];
    float* out = (float*)d_out;

    float*     feats      = (float*)d_ws;                                // 10*64*HW
    float*     om         = feats + (size_t)Bb * Tt * Ff * HW;           // 10*27*HW
    _Float16*  apack_offm = (_Float16*)(om + (size_t)Bb * Tt * 27 * HW); // 2*36*512
    _Float16*  apack_dcn  = apack_offm + 2 * 36 * 512;                   // 4*18*512

    // Weight packing for WMMA A fragments (tiny, once per launch).
    pack_a_f16<<<(2 * 36 * 512 + 255) / 256, 256, 0, stream>>>(offm_w, apack_offm, 2, 36, 27, 1152);
    pack_a_f16<<<(4 * 18 * 512 + 255) / 256, 256, 0, stream>>>(dcn_w,  apack_dcn,  4, 18, 64, 576);

    // Stage 1: features.
    init_conv_prelu<<<(Bb * Tt * Ff * HW + 255) / 256, 256, 0, stream>>>(
        x, init_w, init_b, prelu_a, feats);

    // Stage 2: offset/mask conv (WMMA). One block per (jb, 16-pixel strip).
    offset_conv_wmma<<<Bb * Tt * (HW / 16), 64, 0, stream>>>(feats, apack_offm, offm_b, om);

    // Stage 3: deformable sample + channel GEMM (WMMA + TDM).
    dcn_wmma<<<Bb * Tt * (HW / 16), 128, 0, stream>>>(feats, om, apack_dcn, dcn_b, out);
}
